// GNNBackbone_30270929502807
// MI455X (gfx1250) — compile-verified, matrix operand order checked by reading, not run on hardware
//
#include <hip/hip_runtime.h>

typedef __attribute__((ext_vector_type(16))) _Float16 v16h;
typedef __attribute__((ext_vector_type(8)))  _Float16 v8h;
typedef __attribute__((ext_vector_type(8)))  float    v8f;
typedef __attribute__((ext_vector_type(4)))  float    v4f;

#define NV 1024
#define DV 128
#define EV 32
#define HV 4

// ---------------- WMMA fragment helpers (gfx1250, wave32) ----------------
// 16-bit A 16x32 lane layout (ISA 7.12.2): lane L holds row M=L%16; halfs j=0..7 map
// to K = j + (L>=16?8:0), j=8..15 map to K = j+8 + (L>=16?8:0)  -> two contiguous
// 8-half runs per lane => two 128-bit loads.
__device__ __forceinline__ v8f wmma_f16(v16h a, v16h b, v8f c) {
  return __builtin_amdgcn_wmma_f32_16x16x32_f16(false, a, false, b, (short)0, c, false, false);
}

// wide f16 fragment load (A from f16 row-major, or B from f16 N-major "Bt")
__device__ __forceinline__ v16h load_frag(const _Float16* __restrict__ A, int lda,
                                          int r0, int k0, int lane) {
  const _Float16* p = A + (size_t)(r0 + (lane & 15)) * lda + k0 + ((lane & 16) ? 8 : 0);
  v8h lo = *(const v8h*)p;
  v8h hi = *(const v8h*)(p + 16);
  v16h a;
#pragma unroll
  for (int j = 0; j < 8; ++j) { a[j] = lo[j]; a[j + 8] = hi[j]; }
  return a;
}
// wide f32 fragment load (A from f32 row-major): two contiguous float4 pairs
__device__ __forceinline__ v16h load_frag(const float* __restrict__ A, int lda,
                                          int r0, int k0, int lane) {
  const float* p = A + (size_t)(r0 + (lane & 15)) * lda + k0 + ((lane & 16) ? 8 : 0);
  v4f x0 = *(const v4f*)p;
  v4f x1 = *(const v4f*)(p + 4);
  v4f x2 = *(const v4f*)(p + 16);
  v4f x3 = *(const v4f*)(p + 20);
  v16h a;
#pragma unroll
  for (int j = 0; j < 4; ++j) {
    a[j]      = (_Float16)x0[j];
    a[j + 4]  = (_Float16)x1[j];
    a[j + 8]  = (_Float16)x2[j];
    a[j + 12] = (_Float16)x3[j];
  }
  return a;
}
// B fragment from f32 row-major KxN (activations): dword loads, coalesced across lanes
__device__ __forceinline__ v16h load_bcol(const float* __restrict__ B, int ldb,
                                          int k0, int n0, int lane) {
  const float* p = B + n0 + (lane & 15) + (size_t)(k0 + ((lane & 16) ? 8 : 0)) * ldb;
  v16h b;
#pragma unroll
  for (int j = 0; j < 8; ++j) {
    b[j]     = (_Float16)p[(size_t)j * ldb];
    b[j + 8] = (_Float16)p[(size_t)(j + 16) * ldb];
  }
  return b;
}

// ---------------- generic WMMA GEMM ----------------
// C = act(scale*A@B + bias) (+res).  A: f32/f16 row-major MxK.
// BK==0: B f32 row-major KxN (ldb=N).  BK==1: B f16 N-major "Bt" [N][K] (ldb=K).
template <typename AT, int BK, int ACT, bool BIAS, bool RES>
__global__ void __launch_bounds__(256) k_gemm(const AT* __restrict__ A,
    const void* __restrict__ Bp, const float* __restrict__ bias,
    const float* __restrict__ res, float* __restrict__ C,
    int M, int Nn, int K, int lda, int ldb, int ldc, float scale) {
  int lane = threadIdx.x & 31;
  int t = blockIdx.x * 8 + (threadIdx.x >> 5);
  int tilesM = M >> 4, tilesN = Nn >> 4;
  if (t >= tilesM * tilesN) return;
  int m0 = (t / tilesN) << 4, n0 = (t % tilesN) << 4;
  v8f acc = {};
  for (int k0 = 0; k0 < K; k0 += 32) {
    v16h a = load_frag(A, lda, m0, k0, lane);
    v16h b;
    if (BK == 0) b = load_bcol((const float*)Bp, ldb, k0, n0, lane);
    else         b = load_frag((const _Float16*)Bp, ldb, n0, k0, lane);
    acc = wmma_f16(a, b, acc);
  }
  int col = n0 + (lane & 15);
  int rb  = m0 + ((lane & 16) ? 8 : 0);
  float bv = BIAS ? bias[col] : 0.0f;
#pragma unroll
  for (int r = 0; r < 8; ++r) {
    float v = acc[r] * scale + bv;
    if (ACT == 1) v = fmaxf(v, 0.0f);
    if (ACT == 2) v = v / (1.0f + __expf(-v));       // silu
    if (ACT == 3) v = 1.0f / (1.0f + __expf(-v));    // sigmoid
    if (RES) v += res[(size_t)(rb + r) * ldc + col];
    C[(size_t)(rb + r) * ldc + col] = v;
  }
}

template <typename AT, int BK, int ACT, bool BIAS, bool RES>
static inline void gemm(hipStream_t s, const AT* A, const void* B, const float* bias,
                        const float* res, float* C, int M, int Nn, int K,
                        int lda, int ldb, int ldc, float scale = 1.0f) {
  int tiles = (M / 16) * (Nn / 16);
  k_gemm<AT, BK, ACT, BIAS, RES><<<(tiles + 7) / 8, 256, 0, s>>>(A, B, bias, res, C,
                                                                 M, Nn, K, lda, ldb, ldc, scale);
}

// ---------------- weight transpose-convert: B[KxN] f32 -> Bt[N][K] f16 ----------------
__global__ void k_conv_bt(const float* __restrict__ B, _Float16* __restrict__ Bt,
                          int K, int Nn) {
  int idx = blockIdx.x * 256 + threadIdx.x;
  if (idx >= K * Nn) return;
  int n = idx / K, k = idx % K;
  Bt[idx] = (_Float16)B[(size_t)k * Nn + n];
}
__global__ void k_conv_bt_pad(const float* __restrict__ B, _Float16* __restrict__ Bt,
                              int K, int Nn, int Npad) {
  int idx = blockIdx.x * 256 + threadIdx.x;
  if (idx >= K * Npad) return;
  int n = idx / K, k = idx % K;
  Bt[idx] = (n < Nn) ? (_Float16)B[(size_t)k * Nn + n] : (_Float16)0.0f;
}

// ---------------- fused edge pass ----------------
// Single read of edge_features (134 MB) -> eb (GAT bias), ew/ew2 (GCN/SAGE gates),
// ebias (4-head attention bias). One wave per 16-edge tile; weights pre-converted
// to f16 N-major so every B fragment is two global_load_b128.
__global__ void __launch_bounds__(256) k_edge(const float* __restrict__ ef,
    const _Float16* __restrict__ g1t, const float* __restrict__ gb1,
    const float* __restrict__ gw2, const float* __restrict__ gb2,
    const _Float16* __restrict__ c1t, const float* __restrict__ cb1,
    const float* __restrict__ cw2, const float* __restrict__ cb2,
    const _Float16* __restrict__ s1t, const float* __restrict__ sb1,
    const float* __restrict__ sw2, const float* __restrict__ sb2,
    const _Float16* __restrict__ e1t, const float* __restrict__ e1b,
    const _Float16* __restrict__ e2t, const float* __restrict__ e2b,
    float* __restrict__ eb, float* __restrict__ ew, float* __restrict__ ew2,
    float* __restrict__ ebias) {
  int lane = threadIdx.x & 31, wave = threadIdx.x >> 5;
  size_t t  = (size_t)blockIdx.x * 8 + wave;   // 65536 tiles
  size_t e0 = t * 16;
  __shared__ float    hsh[8][16][EV];          // 16 KB
  __shared__ _Float16 hb[8][16][DV];           // 32 KB

  // A fragment: 16 edges x 32 edge-features (wide f32 loads)
  v16h a = load_frag(ef, EV, (int)e0, 0, lane);

  // E->E (relu) -> E->1 gate MLP, inlined 3x with direct pointers (keeps global_load)
  auto gate = [&](const _Float16* w1t, const float* b1, const float* w2,
                  const float* b2, float* outp, bool sig) {
    v16h bf0 = load_frag(w1t, EV, 0, 0, lane);
    v16h bf1 = load_frag(w1t, EV, 16, 0, lane);
    v8f c0 = {}, c1 = {};
    c0 = wmma_f16(a, bf0, c0);
    c1 = wmma_f16(a, bf1, c1);
    int col = lane & 15, rbl = (lane & 16) ? 8 : 0;
    float x0 = b1[col], x1 = b1[col + 16];
#pragma unroll
    for (int r = 0; r < 8; ++r) {
      hsh[wave][rbl + r][col]      = fmaxf(c0[r] + x0, 0.0f);
      hsh[wave][rbl + r][col + 16] = fmaxf(c1[r] + x1, 0.0f);
    }
    __syncthreads();
    if (lane < 16) {
      float accd = b2[0];
      for (int k = 0; k < EV; ++k) accd += hsh[wave][lane][k] * w2[k];
      outp[e0 + lane] = sig ? 1.0f / (1.0f + __expf(-accd)) : accd;
    }
    __syncthreads();
  };
  gate(g1t, gb1, gw2, gb2, eb,  false);
  gate(c1t, cb1, cw2, cb2, ew,  true);
  gate(s1t, sb1, sw2, sb2, ew2, true);

  // attention edge-bias: silu(A @ ebp1[32x128] + b) @ ebp2[128x4] + b2
#pragma unroll
  for (int nt = 0; nt < 8; ++nt) {
    v16h b = load_frag(e1t, EV, nt * 16, 0, lane);
    v8f c = {};
    c = wmma_f16(a, b, c);
    int col = nt * 16 + (lane & 15), rbl = (lane & 16) ? 8 : 0;
    float bbv = e1b[col];
#pragma unroll
    for (int r = 0; r < 8; ++r) {
      float v = c[r] + bbv;
      hb[wave][rbl + r][col] = (_Float16)(v / (1.0f + __expf(-v)));
    }
  }
  __syncthreads();
  v8f z = {};
#pragma unroll
  for (int k0 = 0; k0 < DV; k0 += 32) {
    v16h a2 = load_frag(&hb[wave][0][0], DV, 0, k0, lane);   // ds_load_b128 x2
    v16h b2 = load_frag(e2t, DV, 0, k0, lane);               // padded Bt[16][128]
    z = wmma_f16(a2, b2, z);
  }
  {
    int col = lane & 15, rbl = (lane & 16) ? 8 : 0;
    if (col < HV) {
      float bv = e2b[col];
#pragma unroll
      for (int r = 0; r < 8; ++r) ebias[(e0 + rbl + r) * HV + col] = z[r] + bv;
    }
  }
}

// ---------------- row-wise kernels ----------------
__global__ void k_rmsnorm(const float* __restrict__ x, const float* __restrict__ w,
                          float* __restrict__ o) {
  __shared__ float rbuf[DV];
  int i = blockIdx.x, tid = threadIdx.x;
  float v = x[(size_t)i * DV + tid];
  rbuf[tid] = v * v;
  __syncthreads();
  for (int s = 64; s > 0; s >>= 1) {
    if (tid < s) rbuf[tid] += rbuf[tid + s];
    __syncthreads();
  }
  float inv = rsqrtf(rbuf[0] * (1.0f / DV) + 1e-6f);
  o[(size_t)i * DV + tid] = v * inv * w[tid];
}

__global__ void k_srcdst(const float* __restrict__ proj, const float* __restrict__ as,
                         const float* __restrict__ at, float* __restrict__ ssrc,
                         float* __restrict__ stgt) {
  __shared__ float r1[DV], r2[DV];
  int i = blockIdx.x, tid = threadIdx.x;
  float p = proj[(size_t)i * DV + tid];
  r1[tid] = p * as[tid];
  r2[tid] = p * at[tid];
  __syncthreads();
  for (int s = 64; s > 0; s >>= 1) {
    if (tid < s) { r1[tid] += r1[tid + s]; r2[tid] += r2[tid + s]; }
    __syncthreads();
  }
  if (tid == 0) { ssrc[i] = r1[0]; stgt[i] = r2[0]; }
}

// GAT row: leaky(s_src+s_tgt+eb), neighbor mask, softmax -> f16 attn
__global__ void __launch_bounds__(256) k_gat_softmax(const float* __restrict__ ssrc,
    const float* __restrict__ stgt, const float* __restrict__ eb,
    const float* __restrict__ adj, _Float16* __restrict__ attn) {
  __shared__ float sv[NV];
  __shared__ float rbuf[256];
  int i = blockIdx.x, tid = threadIdx.x;
  float si = ssrc[i];
  float lmax = -1e30f;
  for (int j = tid; j < NV; j += 256) {
    float v = si + stgt[j] + eb[(size_t)i * NV + j];
    v = (v > 0.0f) ? v : 0.2f * v;
    bool ok = (adj[(size_t)i * NV + j] > 0.0f) || (j == i);
    v = ok ? v : -1e30f;
    sv[j] = v;
    lmax = fmaxf(lmax, v);
  }
  rbuf[tid] = lmax; __syncthreads();
  for (int s = 128; s > 0; s >>= 1) { if (tid < s) rbuf[tid] = fmaxf(rbuf[tid], rbuf[tid + s]); __syncthreads(); }
  float mx = rbuf[0]; __syncthreads();
  float lsum = 0.0f;
  for (int j = tid; j < NV; j += 256) {
    float v = sv[j];
    float e = (v <= -1e30f) ? 0.0f : __expf(v - mx);
    sv[j] = e; lsum += e;
  }
  rbuf[tid] = lsum; __syncthreads();
  for (int s = 128; s > 0; s >>= 1) { if (tid < s) rbuf[tid] += rbuf[tid + s]; __syncthreads(); }
  float inv = rbuf[0] > 0.0f ? 1.0f / rbuf[0] : 0.0f;
  for (int j = tid; j < NV; j += 256)
    attn[(size_t)i * NV + j] = (_Float16)(sv[j] * inv);
}

// GCN/SAGE normalized adjacency: m = (adj*(1+ew)+I)/max(rowsum,1) -> f16
__global__ void __launch_bounds__(256) k_norm_adj(const float* __restrict__ adj,
    const float* __restrict__ ew, _Float16* __restrict__ m) {
  __shared__ float sv[NV];
  __shared__ float rbuf[256];
  int i = blockIdx.x, tid = threadIdx.x;
  float lsum = 0.0f;
  for (int j = tid; j < NV; j += 256) {
    float a = adj[(size_t)i * NV + j] * (1.0f + ew[(size_t)i * NV + j]) + ((j == i) ? 1.0f : 0.0f);
    sv[j] = a; lsum += a;
  }
  rbuf[tid] = lsum; __syncthreads();
  for (int s = 128; s > 0; s >>= 1) { if (tid < s) rbuf[tid] += rbuf[tid + s]; __syncthreads(); }
  float inv = 1.0f / fmaxf(rbuf[0], 1.0f);
  for (int j = tid; j < NV; j += 256)
    m[(size_t)i * NV + j] = (_Float16)(sv[j] * inv);
}

// MHA softmax (mode 0 none, 1 neighbor mask, 2 role mask) -> f16 attn
__global__ void __launch_bounds__(256) k_mha_softmax(const float* __restrict__ S,
    const float* __restrict__ ebias, int head, const float* __restrict__ adj,
    const int* __restrict__ roles, int mode, _Float16* __restrict__ attn) {
  __shared__ float sv[NV];
  __shared__ float rbuf[256];
  int i = blockIdx.x, tid = threadIdx.x;
  int myrole = (mode == 2) ? roles[i] : 0;
  float lmax = -1e30f;
  for (int j = tid; j < NV; j += 256) {
    float v = S[(size_t)i * NV + j];
    if (ebias) v += ebias[((size_t)i * NV + j) * HV + head];
    bool ok = true;
    if (mode == 1) ok = (adj[(size_t)i * NV + j] > 0.0f) || (j == i);
    else if (mode == 2) ok = (roles[j] == myrole);
    v = ok ? v : -1e30f;
    sv[j] = v;
    lmax = fmaxf(lmax, v);
  }
  rbuf[tid] = lmax; __syncthreads();
  for (int s = 128; s > 0; s >>= 1) { if (tid < s) rbuf[tid] = fmaxf(rbuf[tid], rbuf[tid + s]); __syncthreads(); }
  float mx = rbuf[0]; __syncthreads();
  float lsum = 0.0f;
  for (int j = tid; j < NV; j += 256) {
    float v = sv[j];
    float e = (v <= -1e30f) ? 0.0f : __expf(v - mx);
    sv[j] = e; lsum += e;
  }
  rbuf[tid] = lsum; __syncthreads();
  for (int s = 128; s > 0; s >>= 1) { if (tid < s) rbuf[tid] += rbuf[tid + s]; __syncthreads(); }
  float inv = rbuf[0] > 0.0f ? 1.0f / rbuf[0] : 0.0f;  // all-masked row -> zeros (nan_to_num)
  for (int j = tid; j < NV; j += 256)
    attn[(size_t)i * NV + j] = (_Float16)(sv[j] * inv);
}

// QK^T for one head: one WMMA per 16x16 score tile (K = dh = 32)
__global__ void __launch_bounds__(256) k_qk(const float* __restrict__ q,
    const float* __restrict__ kk, float* __restrict__ S, int head, float scale) {
  int lane = threadIdx.x & 31;
  int t = blockIdx.x * 8 + (threadIdx.x >> 5);
  int i0 = (t >> 6) << 4, j0 = (t & 63) << 4;
  v16h a = load_frag(q,  DV, i0, head * 32, lane);
  v16h b = load_frag(kk, DV, j0, head * 32, lane);  // B^T load == A-pattern load
  v8f c = {};
  c = wmma_f16(a, b, c);
  int col = j0 + (lane & 15), rb = i0 + ((lane & 16) ? 8 : 0);
#pragma unroll
  for (int r = 0; r < 8; ++r) S[(size_t)(rb + r) * NV + col] = c[r] * scale;
}

// ---------------- small elementwise kernels ----------------
__global__ void k_concat(const float* __restrict__ x, const float* __restrict__ y,
                         float* __restrict__ o) {
  int idx = blockIdx.x * 256 + threadIdx.x;
  if (idx >= NV * 2 * DV) return;
  int row = idx / (2 * DV), c = idx % (2 * DV);
  o[idx] = (c < DV) ? x[row * DV + c] : y[row * DV + (c - DV)];
}
__global__ void k_gate_add(float* __restrict__ h, const float* __restrict__ g,
                           const float* __restrict__ t, int n) {
  int i = blockIdx.x * 256 + threadIdx.x;
  if (i < n) h[i] += g[i] * t[i];
}
__global__ void k_mul(float* __restrict__ a, const float* __restrict__ b, int n) {
  int i = blockIdx.x * 256 + threadIdx.x;
  if (i < n) a[i] *= b[i];
}

// ---------------- host orchestration ----------------
extern "C" void kernel_launch(void* const* d_in, const int* in_sizes, int n_in,
                              void* d_out, int out_size, void* d_ws, size_t ws_size,
                              hipStream_t stream) {
  (void)in_sizes; (void)n_in; (void)out_size; (void)ws_size;
  const float* feats = (const float*)d_in[0];
  const float* adj   = (const float*)d_in[1];
  const float* ef    = (const float*)d_in[2];
  const int*   roles = (const int*)d_in[3];
  auto F = [&](int i) { return (const float*)d_in[i]; };
  const float *gat_w = F(4), *gat_as = F(5), *gat_at = F(6),
              *g_e1w = F(7), *g_e1b = F(8), *g_e2w = F(9), *g_e2b = F(10),
              *gcn_w = F(11), *gcn_b = F(12),
              *c_e1w = F(13), *c_e1b = F(14), *c_e2w = F(15), *c_e2b = F(16),
              *sage_w = F(17), *sage_b = F(18),
              *s_e1w = F(19), *s_e1b = F(20), *s_e2w = F(21), *s_e2b = F(22),
              *role_q = F(23), *role_k = F(24), *role_v = F(25), *role_o = F(26),
              *nbr_q = F(27), *nbr_k = F(28), *nbr_v = F(29), *nbr_o = F(30),
              *full_q = F(31), *full_k = F(32), *full_v = F(33), *full_o = F(34),
              *ebp1_w = F(35), *ebp1_b = F(36), *ebp2_w = F(37), *ebp2_b = F(38),
              *role_nw = F(39), *nbr_nw = F(40), *full_nw = F(41), *feat_nw = F(42),
              *ffn_nw = F(43), *ft_w = F(44), *ft_b = F(45), *fg_w = F(46), *fg_b = F(47),
              *ffn_gw = F(48), *ffn_uw = F(49), *ffn_dw = F(50);

  float* ws = (float*)d_ws;
  size_t off = 0;
  auto alloc  = [&](size_t n)  { float* p = ws + off; off += n; return p; };
  auto alloch = [&](size_t nh) { _Float16* p = (_Float16*)(ws + off);
                                 off += (nh + 15) / 16 * 8; return p; };
  float* xb0 = alloc(NV * DV);  float* xb1 = alloc(NV * DV);
  float* xb2 = alloc(NV * DV);  float* xb3 = alloc(NV * DV);
  float* qb = alloc(NV * DV);   float* kb = alloc(NV * DV);
  float* vb = alloc(NV * DV);   float* ob = alloc(NV * DV);
  float* cat = alloc((size_t)NV * 2 * DV);
  float* fG = alloc((size_t)NV * 4 * DV);
  float* fU = alloc((size_t)NV * 4 * DV);
  float* ebv    = alloc((size_t)NV * NV);
  float* ewv    = alloc((size_t)NV * NV);
  float* ew2v   = alloc((size_t)NV * NV);
  float* ebiasv = alloc((size_t)NV * NV * HV);
  float* Sbuf   = alloc((size_t)NV * NV);
  float* ssrc = alloc(NV);  float* stgt = alloc(NV);
  _Float16* attnh = alloch((size_t)NV * NV);

  // f16 N-major weight copies ("Bt"): two b128 loads per WMMA B fragment
  auto convbt = [&](const float* B, int K, int Nn) {
    _Float16* Bt = alloch((size_t)K * Nn);
    k_conv_bt<<<(K * Nn + 255) / 256, 256, 0, stream>>>(B, Bt, K, Nn);
    return Bt;
  };
  _Float16* gat_wT  = convbt(gat_w, DV, DV);
  _Float16* gcn_wT  = convbt(gcn_w, DV, DV);
  _Float16* sage_wT = convbt(sage_w, 2 * DV, DV);
  _Float16* role_qT = convbt(role_q, DV, DV); _Float16* role_kT = convbt(role_k, DV, DV);
  _Float16* role_vT = convbt(role_v, DV, DV); _Float16* role_oT = convbt(role_o, DV, DV);
  _Float16* nbr_qT  = convbt(nbr_q, DV, DV);  _Float16* nbr_kT  = convbt(nbr_k, DV, DV);
  _Float16* nbr_vT  = convbt(nbr_v, DV, DV);  _Float16* nbr_oT  = convbt(nbr_o, DV, DV);
  _Float16* full_qT = convbt(full_q, DV, DV); _Float16* full_kT = convbt(full_k, DV, DV);
  _Float16* full_vT = convbt(full_v, DV, DV); _Float16* full_oT = convbt(full_o, DV, DV);
  _Float16* ft_wT   = convbt(ft_w, DV, DV);
  _Float16* fg_wT   = convbt(fg_w, 2 * DV, DV);
  _Float16* ffn_gwT = convbt(ffn_gw, DV, 4 * DV);
  _Float16* ffn_uwT = convbt(ffn_uw, DV, 4 * DV);
  _Float16* ffn_dwT = convbt(ffn_dw, 4 * DV, DV);
  _Float16* g_e1T   = convbt(g_e1w, EV, EV);
  _Float16* c_e1T   = convbt(c_e1w, EV, EV);
  _Float16* s_e1T   = convbt(s_e1w, EV, EV);
  _Float16* ebp1T   = convbt(ebp1_w, EV, DV);
  _Float16* ebp2T   = alloch((size_t)DV * 16);          // [16][128], cols 4..15 zero
  k_conv_bt_pad<<<(DV * 16 + 255) / 256, 256, 0, stream>>>(ebp2_w, ebp2T, DV, HV, 16);

  // 1. fused edge pass (single read of 134 MB edge_features)
  k_edge<<<8192, 256, 0, stream>>>(ef, g_e1T, g_e1b, g_e2w, g_e2b,
      c_e1T, c_e1b, c_e2w, c_e2b, s_e1T, s_e1b, s_e2w, s_e2b,
      ebp1T, ebp1_b, ebp2T, ebp2_b, ebv, ewv, ew2v, ebiasv);

  // 2. DenseGAT
  gemm<float, 1, 0, false, false>(stream, feats, gat_wT, nullptr, nullptr, xb1, NV, DV, DV, DV, DV, DV);
  k_srcdst<<<NV, 128, 0, stream>>>(xb1, gat_as, gat_at, ssrc, stgt);
  k_gat_softmax<<<NV, 256, 0, stream>>>(ssrc, stgt, ebv, adj, attnh);
  gemm<_Float16, 0, 1, false, false>(stream, attnh, xb1, nullptr, nullptr, xb0, NV, DV, NV, NV, DV, DV);

  // 3. DenseGCN
  k_norm_adj<<<NV, 256, 0, stream>>>(adj, ewv, attnh);
  gemm<_Float16, 0, 0, false, false>(stream, attnh, xb0, nullptr, nullptr, xb1, NV, DV, NV, NV, DV, DV);
  gemm<float, 1, 1, true, false>(stream, xb1, gcn_wT, gcn_b, nullptr, xb2, NV, DV, DV, DV, DV, DV);

  // 4. DenseGraphSAGE (1/deg folded into f16 adjacency)
  k_norm_adj<<<NV, 256, 0, stream>>>(adj, ew2v, attnh);
  gemm<_Float16, 0, 0, false, false>(stream, attnh, xb2, nullptr, nullptr, xb1, NV, DV, NV, NV, DV, DV);
  k_concat<<<(NV * 2 * DV + 255) / 256, 256, 0, stream>>>(xb2, xb1, cat);
  gemm<float, 1, 1, true, false>(stream, cat, sage_wT, sage_b, nullptr, xb0, NV, DV, 2 * DV, 2 * DV, 2 * DV, DV);

  // 5-7. three relational attention blocks
  const float scl = 0.17677669529663687f;  // 32^-0.5
  auto attn_block = [&](const float* nw, const _Float16* wqT, const _Float16* wkT,
                        const _Float16* wvT, const _Float16* woT, int mode, const float* ebp) {
    k_rmsnorm<<<NV, 128, 0, stream>>>(xb0, nw, xb1);
    gemm<float, 1, 0, false, false>(stream, xb1, wqT, nullptr, nullptr, qb, NV, DV, DV, DV, DV, DV);
    gemm<float, 1, 0, false, false>(stream, xb1, wkT, nullptr, nullptr, kb, NV, DV, DV, DV, DV, DV);
    gemm<float, 1, 0, false, false>(stream, xb1, wvT, nullptr, nullptr, vb, NV, DV, DV, DV, DV, DV);
    for (int h = 0; h < HV; ++h) {
      k_qk<<<512, 256, 0, stream>>>(qb, kb, Sbuf, h, scl);
      k_mha_softmax<<<NV, 256, 0, stream>>>(Sbuf, ebp, h, adj, roles, mode, attnh);
      gemm<_Float16, 0, 0, false, false>(stream, attnh, vb + h * 32, nullptr, nullptr,
                                         ob + h * 32, NV, 32, NV, NV, DV, DV);
    }
    gemm<float, 1, 0, false, true>(stream, ob, woT, nullptr, xb0, xb0, NV, DV, DV, DV, DV, DV);
  };
  attn_block(role_nw, role_qT, role_kT, role_vT, role_oT, 2, nullptr);
  attn_block(nbr_nw,  nbr_qT,  nbr_kT,  nbr_vT,  nbr_oT,  1, ebiasv);
  attn_block(full_nw, full_qT, full_kT, full_vT, full_oT, 0, nullptr);

  // 8. feature mixing: h += sigmoid([fn,t]@fg_w+b) * silu(fn@ft_w+b)
  k_rmsnorm<<<NV, 128, 0, stream>>>(xb0, feat_nw, xb1);
  gemm<float, 1, 2, true, false>(stream, xb1, ft_wT, ft_b, nullptr, xb2, NV, DV, DV, DV, DV, DV);
  k_concat<<<(NV * 2 * DV + 255) / 256, 256, 0, stream>>>(xb1, xb2, cat);
  gemm<float, 1, 3, true, false>(stream, cat, fg_wT, fg_b, nullptr, xb3, NV, DV, 2 * DV, 2 * DV, 2 * DV, DV);
  k_gate_add<<<(NV * DV + 255) / 256, 256, 0, stream>>>(xb0, xb3, xb2, NV * DV);

  // 9. gated FFN, final residual GEMM writes d_out
  k_rmsnorm<<<NV, 128, 0, stream>>>(xb0, ffn_nw, xb1);
  gemm<float, 1, 2, false, false>(stream, xb1, ffn_gwT, nullptr, nullptr, fG, NV, 4 * DV, DV, DV, DV, 4 * DV);
  gemm<float, 1, 0, false, false>(stream, xb1, ffn_uwT, nullptr, nullptr, fU, NV, 4 * DV, DV, DV, DV, 4 * DV);
  k_mul<<<(NV * 4 * DV + 255) / 256, 256, 0, stream>>>(fG, fU, NV * 4 * DV);
  gemm<float, 1, 0, false, true>(stream, fG, ffn_dwT, nullptr, xb0, (float*)d_out,
                                 NV, DV, 4 * DV, 4 * DV, 4 * DV, DV);
}
